// VectorQuantizer_42262478192886
// MI455X (gfx1250) — matrix operand order repaired and checked by reading, not compiled, not run on hardware
//
#include <hip/hip_runtime.h>

// ---------------- problem constants (from reference) ----------------
#define BATCH   32
#define DDIM    256      // d (contraction dim)
#define NSEQ    4096     // n
#define KCODES  512      // codebook size
#define MTOTAL  (BATCH*NSEQ)          // 131072 rows
#define QELEMS  (MTOTAL*DDIM)         // 33554432 quantized elements
#define IDX_OFF QELEMS
#define LOSS_OFF (QELEMS + MTOTAL)

// ---------------- tiling ----------------
#define BM   64          // rows per workgroup
#define BN   64          // codebook columns staged per LDS chunk
#define AST  264         // padded LDS/ws row stride in ushorts (264*2B = 33 x 16B, conflict-free b128)

typedef __bf16 bf16_t;
typedef bf16_t v16bf __attribute__((ext_vector_type(16)));
typedef float  v8f   __attribute__((ext_vector_type(8)));
typedef unsigned int u32x4 __attribute__((ext_vector_type(4)));
typedef int    v4i   __attribute__((ext_vector_type(4)));
typedef float  f32x2 __attribute__((ext_vector_type(2)));
typedef float  f32x4 __attribute__((ext_vector_type(4)));
typedef int    i32x4 __attribute__((ext_vector_type(4)));
typedef unsigned short u16;
typedef unsigned int   u32;

union FragBF { u32x4 q[2]; v16bf v; };

// Async global->LDS DMA path (ASYNCcnt), compile-safe via __has_builtin.
#if defined(__gfx1250__) && __has_builtin(__builtin_amdgcn_global_load_async_to_lds_b128)
#define ASYNC_LDS 1
#define AS1 __attribute__((address_space(1)))
#define AS3 __attribute__((address_space(3)))
__device__ __forceinline__ void async_copy16(const void* g, void* l) {
  // signature: (v4i AS1* global_src, v4i AS3* lds_dst, imm offset, imm cpol)
  __builtin_amdgcn_global_load_async_to_lds_b128((AS1 v4i*)g, (AS3 v4i*)l, 0, 0);
}
#endif

__device__ __forceinline__ u16 f2bf(float f) {
  u32 u = __float_as_uint(f);
  u += 0x7FFFu + ((u >> 16) & 1u);          // round-to-nearest-even
  return (u16)(u >> 16);
}
__device__ __forceinline__ float bf2f(u16 h) {
  return __uint_as_float(((u32)h) << 16);
}
__device__ __forceinline__ u32 packbf(float a, float b) {  // a -> low d, b -> high d
  return (u32)f2bf(a) | ((u32)f2bf(b) << 16);
}

// =====================================================================
// Kernel 1: split codebook into bf16 hi/lo (k-major, padded stride) and
// compute exact ||e_k||^2.  embedding layout: [d=256][K=512] row-major.
// =====================================================================
__global__ __launch_bounds__(256) void vq_prep(const float* __restrict__ emb,
                                               u16* __restrict__ ehi,
                                               u16* __restrict__ elo,
                                               float* __restrict__ enorm) {
  const int k = blockIdx.x;      // 0..511
  const int d = threadIdx.x;     // 0..255
  float f  = emb[(size_t)d * KCODES + k];
  u16 h    = f2bf(f);
  u16 l    = f2bf(f - bf2f(h));
  ehi[k * AST + d] = h;
  elo[k * AST + d] = l;
  __shared__ float red[256];
  red[d] = f * f;
  __syncthreads();
  for (int s = 128; s > 0; s >>= 1) {
    if (d < s) red[d] += red[d + s];
    __syncthreads();
  }
  if (d == 0) enorm[k] = red[0];
}

// =====================================================================
// Kernel 2: distances + argmin via split-bf16 WMMA.
//   dist(m,k) = ||e_k||^2 - 2 * x_m . e_k     (||x_m||^2 constant per row)
//   dot computed as hi*hi + hi*lo + lo*hi  (~fp32 accuracy), f32 accum.
// 256 threads = 8 waves; wave (wm,wn): wm owns a 16-row M tile, wn owns
// two 16-col tiles of each 64-col N chunk -> dual accumulators so the
// WMMA chains interleave (no dependent back-to-back WMMAs) and each A
// fragment is loaded from LDS once per d-chunk.
// =====================================================================
__global__ __launch_bounds__(256) void vq_argmin(const float* __restrict__ inp,
                                                 const u16* __restrict__ ehi,
                                                 const u16* __restrict__ elo,
                                                 const float* __restrict__ enorm,
                                                 int*   __restrict__ widx,
                                                 float* __restrict__ out_idx) {
  __shared__ __align__(16) u16 sAhi[BM * AST];
  __shared__ __align__(16) u16 sAlo[BM * AST];
  __shared__ __align__(16) u16 sBhi[BN * AST];
  __shared__ __align__(16) u16 sBlo[BN * AST];
  __shared__ float redD[4][16][2];
  __shared__ int   redI[4][16][2];

  const int tid  = threadIdx.x;
  const int lane = tid & 31;
  const int wave = tid >> 5;
  const int wm   = wave >> 1;
  const int wn   = wave & 1;
  const int l15  = lane & 15;
  const int half = lane >> 4;        // 0: lanes 0-15, 1: lanes 16-31

  const int m0  = blockIdx.x * BM;   // 64 rows, all within one batch b
  const int b   = m0 >> 12;          // m = b*4096 + nj
  const int nj0 = m0 & (NSEQ - 1);
  const float* aBase = inp + (size_t)b * DDIM * NSEQ + nj0;

  // ---- stage A tile [64 rows][256 d] into LDS with hi/lo bf16 split ----
  // thread: col pair (2cp, 2cp+1), d pair (d, d+1); b64 loads, packed b32 stores
  {
    const int cp = tid & 31;         // cols 2cp, 2cp+1 (coalesced 256B per wave)
    const int dp = tid >> 5;         // 0..7
    for (int d = 2 * dp; d < DDIM; d += 16) {
      f32x2 f0 = *(const f32x2*)&aBase[(size_t)d * NSEQ + 2 * cp];        // d
      f32x2 f1 = *(const f32x2*)&aBase[(size_t)(d + 1) * NSEQ + 2 * cp];  // d+1
      const int c0 = 2 * cp, c1 = 2 * cp + 1;
      float h0x = bf2f(f2bf(f0.x)), h1x = bf2f(f2bf(f1.x));
      float h0y = bf2f(f2bf(f0.y)), h1y = bf2f(f2bf(f1.y));
      *(u32*)&sAhi[c0 * AST + d] = packbf(f0.x, f1.x);
      *(u32*)&sAhi[c1 * AST + d] = packbf(f0.y, f1.y);
      *(u32*)&sAlo[c0 * AST + d] = packbf(f0.x - h0x, f1.x - h1x);
      *(u32*)&sAlo[c1 * AST + d] = packbf(f0.y - h0y, f1.y - h1y);
    }
  }

  float bestD0[8], bestD1[8];
  int   bestI0[8], bestI1[8];
#pragma unroll
  for (int v = 0; v < 8; ++v) {
    bestD0[v] = 3.0e38f; bestI0[v] = 0;
    bestD1[v] = 3.0e38f; bestI1[v] = 0;
  }

  const int aRow  = (wm * 16 + l15) * AST;          // A fragment row base
  const int bRow0 = (wn * 32 + l15) * AST;          // tile tl0 = wn*2
  const int bRow1 = bRow0 + 16 * AST;               // tile tl1 = wn*2+1

  for (int c = 0; c < KCODES / BN; ++c) {
    __syncthreads();                          // prev chunk consumed (c=0: A staged)
    // ---- stage B chunk: contiguous (ws is k-major, stride AST) ----
    {
      const u32x4* sh = (const u32x4*)(ehi + (size_t)c * BN * AST);
      const u32x4* sl = (const u32x4*)(elo + (size_t)c * BN * AST);
      u32x4* dh = (u32x4*)sBhi;
      u32x4* dl = (u32x4*)sBlo;
      const int nvec = BN * AST / 8;          // 2112 x 16B
#ifdef ASYNC_LDS
      for (int i = tid; i < nvec; i += 256) {
        async_copy16(sh + i, dh + i);
        async_copy16(sl + i, dl + i);
      }
#if __has_builtin(__builtin_amdgcn_s_wait_asynccnt)
      __builtin_amdgcn_s_wait_asynccnt(0);
#else
      asm volatile("s_wait_asynccnt 0" ::: "memory");
#endif
#else
      for (int i = tid; i < nvec; i += 256) { dh[i] = sh[i]; dl[i] = sl[i]; }
#endif
    }
    __syncthreads();

    v8f acc0 = {0.f, 0.f, 0.f, 0.f, 0.f, 0.f, 0.f, 0.f};
    v8f acc1 = {0.f, 0.f, 0.f, 0.f, 0.f, 0.f, 0.f, 0.f};
#pragma unroll
    for (int dk = 0; dk < DDIM; dk += 32) {
      // A 16x32 bf16 frag: lane half selects K +0/+8; b128 pairs at d+{0,16}
      const int ao  = aRow + dk + half * 8;
      // B 32x16 bf16 frag: lane half selects K +0/+8; b128 pairs at d+{0,16}
      const int bo0 = bRow0 + dk + half * 8;
      const int bo1 = bRow1 + dk + half * 8;
      FragBF ah, al, bh0, bl0, bh1, bl1;
      ah.q[0]  = *(const u32x4*)&sAhi[ao];
      ah.q[1]  = *(const u32x4*)&sAhi[ao + 16];
      al.q[0]  = *(const u32x4*)&sAlo[ao];
      al.q[1]  = *(const u32x4*)&sAlo[ao + 16];
      bh0.q[0] = *(const u32x4*)&sBhi[bo0];
      bh0.q[1] = *(const u32x4*)&sBhi[bo0 + 16];
      bl0.q[0] = *(const u32x4*)&sBlo[bo0];
      bl0.q[1] = *(const u32x4*)&sBlo[bo0 + 16];
      bh1.q[0] = *(const u32x4*)&sBhi[bo1];
      bh1.q[1] = *(const u32x4*)&sBhi[bo1 + 16];
      bl1.q[0] = *(const u32x4*)&sBlo[bo1];
      bl1.q[1] = *(const u32x4*)&sBlo[bo1 + 16];
      // interleave two independent accumulation chains
      acc0 = __builtin_amdgcn_wmma_f32_16x16x32_bf16(false, ah.v, false, bh0.v,
                                                     (short)0, acc0, false, false);
      acc1 = __builtin_amdgcn_wmma_f32_16x16x32_bf16(false, ah.v, false, bh1.v,
                                                     (short)0, acc1, false, false);
      acc0 = __builtin_amdgcn_wmma_f32_16x16x32_bf16(false, ah.v, false, bl0.v,
                                                     (short)0, acc0, false, false);
      acc1 = __builtin_amdgcn_wmma_f32_16x16x32_bf16(false, ah.v, false, bl1.v,
                                                     (short)0, acc1, false, false);
      acc0 = __builtin_amdgcn_wmma_f32_16x16x32_bf16(false, al.v, false, bh0.v,
                                                     (short)0, acc0, false, false);
      acc1 = __builtin_amdgcn_wmma_f32_16x16x32_bf16(false, al.v, false, bh1.v,
                                                     (short)0, acc1, false, false);
    }
    const int kcol0 = c * BN + wn * 32 + l15;
    const int kcol1 = kcol0 + 16;
    const float en0 = enorm[kcol0];
    const float en1 = enorm[kcol1];
#pragma unroll
    for (int v = 0; v < 8; ++v) {
      float d0 = en0 - 2.0f * acc0[v];        // lane holds col l15, rows v+8*half
      float d1 = en1 - 2.0f * acc1[v];
      if (d0 < bestD0[v]) { bestD0[v] = d0; bestI0[v] = kcol0; }
      if (d1 < bestD1[v]) { bestD1[v] = d1; bestI1[v] = kcol1; }
    }
  }

  // ---- merge the two per-lane candidates, then argmin across 16 lanes ----
#pragma unroll
  for (int v = 0; v < 8; ++v) {
    float bd = bestD0[v];
    int   bi = bestI0[v];
    if (bestD1[v] < bd || (bestD1[v] == bd && bestI1[v] < bi)) { bd = bestD1[v]; bi = bestI1[v]; }
    for (int off = 8; off > 0; off >>= 1) {
      float od = __shfl_xor(bd, off, 16);
      int   oi = __shfl_xor(bi, off, 16);
      if (od < bd || (od == bd && oi < bi)) { bd = od; bi = oi; }
    }
    if (l15 == 0) {
      const int r = v + 8 * half;             // C-tile row
      redD[wm][r][wn] = bd;
      redI[wm][r][wn] = bi;
    }
  }
  __syncthreads();
  // ---- combine the two N-half waves per row, emit index ----
  if (tid < 64) {
    const int w = tid >> 4, r = tid & 15;
    float d0 = redD[w][r][0]; int i0 = redI[w][r][0];
    float d1 = redD[w][r][1]; int i1 = redI[w][r][1];
    int fi = (d1 < d0 || (d1 == d0 && i1 < i0)) ? i1 : i0;
    widx[m0 + tid]    = fi;
    out_idx[m0 + tid] = (float)fi;
  }
}

// =====================================================================
// Kernel 3: gather fp32 codes -> quantized output (4-wide streaming),
// per-block partial of sum((q - x)^2)  (deterministic order).
// =====================================================================
__global__ __launch_bounds__(256) void vq_quant(const float* __restrict__ inp,
                                                const float* __restrict__ emb,
                                                const int* __restrict__ widx,
                                                float* __restrict__ outq,
                                                float* __restrict__ part) {
  const int VQ = QELEMS / 4;
  float s = 0.f;
  for (int ov = blockIdx.x * 256 + threadIdx.x; ov < VQ; ov += gridDim.x * 256) {
    const int o  = ov * 4;
    const int nj = o & (NSEQ - 1);
    const int d  = (o >> 12) & (DDIM - 1);
    const int b  = o >> 20;
    const i32x4 kk = *(const i32x4*)&widx[(b << 12) + nj];
    const f32x4 x  = *(const f32x4*)&inp[o];
    f32x4 q;
    q.x = emb[d * KCODES + kk.x];             // fp32 codebook, L2-resident
    q.y = emb[d * KCODES + kk.y];
    q.z = emb[d * KCODES + kk.z];
    q.w = emb[d * KCODES + kk.w];
    *(f32x4*)&outq[o] = q;                    // quantized_st == quantized (fwd value)
    const f32x4 df = q - x;
    s += df.x * df.x + df.y * df.y + df.z * df.z + df.w * df.w;
  }
  __shared__ float red[256];
  red[threadIdx.x] = s;
  __syncthreads();
  for (int st = 128; st > 0; st >>= 1) {
    if (threadIdx.x < st) red[threadIdx.x] += red[threadIdx.x + st];
    __syncthreads();
  }
  if (threadIdx.x == 0) part[blockIdx.x] = red[0];
}

// =====================================================================
// Kernel 4: final deterministic sum -> mean -> loss scalar.
// =====================================================================
__global__ __launch_bounds__(256) void vq_loss(const float* __restrict__ part,
                                               float* __restrict__ loss) {
  float s = 0.f;
  for (int i = 0; i < 8; ++i) s += part[threadIdx.x + i * 256];
  __shared__ float red[256];
  red[threadIdx.x] = s;
  __syncthreads();
  for (int st = 128; st > 0; st >>= 1) {
    if (threadIdx.x < st) red[threadIdx.x] += red[threadIdx.x + st];
    __syncthreads();
  }
  if (threadIdx.x == 0) *loss = red[0] * (1.0f / (float)QELEMS);  // COMMITMENT_COST=1
}

// =====================================================================
extern "C" void kernel_launch(void* const* d_in, const int* in_sizes, int n_in,
                              void* d_out, int out_size, void* d_ws, size_t ws_size,
                              hipStream_t stream) {
  const float* inp = (const float*)d_in[0];   // [32,256,4096] fp32
  const float* emb = (const float*)d_in[1];   // [256,512] fp32

  float* out      = (float*)d_out;
  float* out_idx  = out + IDX_OFF;            // indices as float
  float* out_loss = out + LOSS_OFF;

  // workspace layout (bytes, 16B-aligned chunks)
  char* ws = (char*)d_ws;
  u16*   ehi   = (u16*)(ws);                              // 512*264*2 = 270336
  u16*   elo   = (u16*)(ws + 270336);                     // 270336
  float* enorm = (float*)(ws + 540672);                   // 2048
  int*   widx  = (int*)(ws + 542720);                     // 131072*4 = 524288
  float* part  = (float*)(ws + 1067008);                  // 2048*4

  vq_prep  <<<KCODES,        256, 0, stream>>>(emb, ehi, elo, enorm);
  vq_argmin<<<MTOTAL / BM,   256, 0, stream>>>(inp, ehi, elo, enorm, widx, out_idx);
  vq_quant <<<2048,          256, 0, stream>>>(inp, emb, widx, out, part);
  vq_loss  <<<1,             256, 0, stream>>>(part, out_loss);
}
